// DeepseekV4Indexer_13761075216599
// MI455X (gfx1250) — matrix operand order repaired and checked by reading, not compiled, not run on hardware
//
#include <hip/hip_runtime.h>
#include <hip/hip_bf16.h>

// ---------------------------------------------------------------------------
// DeepseekV4 indexer for MI455X (gfx1250): bf16 WMMA GEMMs + RoPE + scores +
// bitonic top-k.  v_wmma_f32_16x16x32_bf16 everywhere, fp32 accumulate,
// register-blocked 32x64 wave tiles on the big GEMM, async-to-LDS staging.
// ---------------------------------------------------------------------------

#define S_LEN 2048
#define HID_  2048
#define H_    32
#define D_    128
#define RD_   64
#define NQ_   (H_ * D_)   // 4096
#define TOPK_ 512

typedef __attribute__((ext_vector_type(16))) __bf16 v16bf;
typedef __attribute__((ext_vector_type(8)))  __bf16 v8bf;
typedef __attribute__((ext_vector_type(8)))  float  v8f;

__device__ __forceinline__ __bf16 f2bf(float f) {
  unsigned u = __float_as_uint(f);
  u += 0x7FFFu + ((u >> 16) & 1u);          // round-to-nearest-even
  unsigned short h = (unsigned short)(u >> 16);
  return __builtin_bit_cast(__bf16, h);
}

// A-fragment: 16 contiguous K values per lane (lane-half picks K 0-15 / 16-31)
__device__ __forceinline__ v16bf ldA(const __bf16* p) {
  v8bf lo = *(const v8bf*)p;
  v8bf hi = *(const v8bf*)(p + 8);
  v16bf r;
#pragma unroll
  for (int t = 0; t < 8; ++t) { r[t] = lo[t]; r[t + 8] = hi[t]; }
  return r;
}
// B-fragment: 16 contiguous K values per lane (lane-half picks K 0-15 / 16-31)
__device__ __forceinline__ v16bf ldB(const __bf16* p) {
  v8bf lo = *(const v8bf*)p;
  v8bf hi = *(const v8bf*)(p + 8);
  v16bf r;
#pragma unroll
  for (int t = 0; t < 8; ++t) { r[t] = lo[t]; r[t + 8] = hi[t]; }
  return r;
}

// ---------------------------------------------------------------------------
// Elementwise fp32 -> bf16
// ---------------------------------------------------------------------------
__global__ void k_f32_to_bf16(const float* __restrict__ src,
                              __bf16* __restrict__ dst, size_t n) {
  size_t i = (size_t)blockIdx.x * blockDim.x + threadIdx.x;
  if (i < n) dst[i] = f2bf(src[i]);
}

// Transpose (Kd x Nd row-major) -> (Nd x Kd row-major) with bf16 convert.
__global__ void k_transpose_bf16(const float* __restrict__ src,
                                 __bf16* __restrict__ dst, int Kd, int Nd) {
  size_t i = (size_t)blockIdx.x * blockDim.x + threadIdx.x;
  size_t n = (size_t)Kd * Nd;
  if (i >= n) return;
  int nn = (int)(i % Nd);
  int kk = (int)(i / Nd);
  dst[(size_t)nn * Kd + kk] = f2bf(src[i]);
}

// ---------------------------------------------------------------------------
// Big GEMM (q-projection): C[MxN] = A[MxK] * Bt[NxK]^T, bf16 in / f32 out.
// Per-wave tile 32(M) x 64(N): 8 accumulators, A reused x4, B reused x2
// -> 8 WMMA per 12 b128 loads.  Block = 8 waves (4M x 2N) = 128x128 macro.
// ---------------------------------------------------------------------------
__global__ __launch_bounds__(256)
void k_wmma_gemm_q(const __bf16* __restrict__ A,
                   const __bf16* __restrict__ Bt,
                   float* __restrict__ C,
                   int M, int N, int K) {
  const int lane = threadIdx.x & 31;
  const int wave = threadIdx.x >> 5;
  const int wm = wave >> 1;                   // 0..3
  const int wn = wave & 1;                    // 0..1
  const int mbase = blockIdx.x * 128 + wm * 32;
  const int nbase = blockIdx.y * 128 + wn * 64;

  const int khalfA = (lane >> 4) * 16;
  const int kselB  = (lane >> 4) * 16;
  const int mrow   = mbase + (lane & 15);
  const int ncol   = nbase + (lane & 15);

  const __bf16* ap0 = A + (size_t)mrow * K + khalfA;
  const __bf16* ap1 = ap0 + (size_t)16 * K;
  const __bf16* bp0 = Bt + (size_t)ncol * K + kselB;
  const __bf16* bp1 = bp0 + (size_t)16 * K;
  const __bf16* bp2 = bp0 + (size_t)32 * K;
  const __bf16* bp3 = bp0 + (size_t)48 * K;

  v8f acc[2][4] = {};
  for (int kk = 0; kk < K; kk += 32) {
    __builtin_prefetch(ap0 + kk + 256, 0, 1);   // global_prefetch_b8
    v16bf a0 = ldA(ap0 + kk);
    v16bf a1 = ldA(ap1 + kk);
    v16bf b0 = ldB(bp0 + kk);
    v16bf b1 = ldB(bp1 + kk);
    v16bf b2 = ldB(bp2 + kk);
    v16bf b3 = ldB(bp3 + kk);
    acc[0][0] = __builtin_amdgcn_wmma_f32_16x16x32_bf16(false, a0, false, b0, (short)0, acc[0][0], false, false);
    acc[0][1] = __builtin_amdgcn_wmma_f32_16x16x32_bf16(false, a0, false, b1, (short)0, acc[0][1], false, false);
    acc[0][2] = __builtin_amdgcn_wmma_f32_16x16x32_bf16(false, a0, false, b2, (short)0, acc[0][2], false, false);
    acc[0][3] = __builtin_amdgcn_wmma_f32_16x16x32_bf16(false, a0, false, b3, (short)0, acc[0][3], false, false);
    acc[1][0] = __builtin_amdgcn_wmma_f32_16x16x32_bf16(false, a1, false, b0, (short)0, acc[1][0], false, false);
    acc[1][1] = __builtin_amdgcn_wmma_f32_16x16x32_bf16(false, a1, false, b1, (short)0, acc[1][1], false, false);
    acc[1][2] = __builtin_amdgcn_wmma_f32_16x16x32_bf16(false, a1, false, b2, (short)0, acc[1][2], false, false);
    acc[1][3] = __builtin_amdgcn_wmma_f32_16x16x32_bf16(false, a1, false, b3, (short)0, acc[1][3], false, false);
  }

  const int mofs = (lane >> 4) * 8;
#pragma unroll
  for (int m = 0; m < 2; ++m)
#pragma unroll
    for (int n = 0; n < 4; ++n)
#pragma unroll
      for (int v = 0; v < 8; ++v)
        C[(size_t)(mbase + m * 16 + mofs + v) * N + (ncol + n * 16)] = acc[m][n][v];
}

// ---------------------------------------------------------------------------
// Generic small GEMM (k / w projections): one 16x16 tile per wave.
// ---------------------------------------------------------------------------
__global__ __launch_bounds__(256)
void k_wmma_gemm_bf16nt(const __bf16* __restrict__ A,
                        const __bf16* __restrict__ Bt,
                        float* __restrict__ C,
                        int M, int N, int K, float scale) {
  const int lane = threadIdx.x & 31;
  const int wave = threadIdx.x >> 5;
  const int wm = wave >> 2;
  const int wn = wave & 3;
  const int mbase = blockIdx.x * 32 + wm * 16;
  const int nbase = blockIdx.y * 64 + wn * 16;
  if (mbase >= M || nbase >= N) return;

  const int arow   = mbase + (lane & 15);
  const int khalfA = (lane >> 4) * 16;
  const int bcol   = nbase + (lane & 15);
  const int kselB  = (lane >> 4) * 16;

  const __bf16* aptr = A  + (size_t)arow * K + khalfA;
  const __bf16* bptr = Bt + (size_t)bcol * K + kselB;

  v8f acc = {};
  for (int kk = 0; kk < K; kk += 32) {
    v16bf a = ldA(aptr + kk);
    v16bf b = ldB(bptr + kk);
    acc = __builtin_amdgcn_wmma_f32_16x16x32_bf16(
        false, a, false, b, (short)0, acc, false, false);
  }

  const int mofs = (lane >> 4) * 8;
#pragma unroll
  for (int v = 0; v < 8; ++v)
    C[(size_t)(mbase + mofs + v) * N + bcol] = acc[v] * scale;
}

// ---------------------------------------------------------------------------
// Interleaved RoPE on the last RDd dims + fp32 -> bf16.  x: (S, Hn, Dd)
// ---------------------------------------------------------------------------
__global__ void k_rope_bf16(const float* __restrict__ x,
                            const float* __restrict__ cosb,
                            const float* __restrict__ sinb,
                            __bf16* __restrict__ out,
                            int Hn, int Dd, int RDd) {
  size_t tid = (size_t)blockIdx.x * blockDim.x + threadIdx.x;
  const int halfD = Dd >> 1;
  size_t npairs = (size_t)S_LEN * Hn * halfD;
  if (tid >= npairs) return;
  int p = (int)(tid % halfD);
  size_t ih = tid / halfD;
  int h = (int)(ih % Hn);
  int i = (int)(ih / Hn);
  size_t base = ((size_t)i * Hn + h) * Dd;
  int d0 = 2 * p;
  float x1 = x[base + d0];
  float x2 = x[base + d0 + 1];
  float o1 = x1, o2 = x2;
  if (d0 >= Dd - RDd) {
    int rr = (d0 - (Dd - RDd)) >> 1;
    float c = cosb[(size_t)i * (RDd >> 1) + rr];
    float s = sinb[(size_t)i * (RDd >> 1) + rr];
    o1 = x1 * c - x2 * s;
    o2 = x2 * c + x1 * s;
  }
  out[base + d0]     = f2bf(o1);
  out[base + d0 + 1] = f2bf(o2);
}

// ---------------------------------------------------------------------------
// scores(i,j) = causal_mask( sum_h relu( (q_ih . k_j) / sqrt(D) ) * w_ih )
// B-fragments (key tile) hoisted into 32 VGPRs, reused across all 32 heads.
// w[i,h] tile staged via gfx1250 async copy: global_load_async_to_lds_b128.
// ---------------------------------------------------------------------------
__global__ __launch_bounds__(256)
void k_scores(const __bf16* __restrict__ Q,   // (S, H, D)
              const __bf16* __restrict__ Kb,  // (S, D)  == B^T, K-contiguous
              const float* __restrict__ W,    // (S, H)
              float* __restrict__ Sc,         // (S, S)
              float inv_sqrt_d) {
  __builtin_amdgcn_s_wait_tensorcnt(0);

  __shared__ float wbuf[32 * H_];
  const int lane = threadIdx.x & 31;
  const int wave = threadIdx.x >> 5;
  const int wm = wave >> 2, wn = wave & 3;
  const int iblk  = blockIdx.x * 32;
  const int ibase = iblk + wm * 16;
  const int jbase = blockIdx.y * 64 + wn * 16;

  // Async-stage the 4KB contiguous w-tile (rows iblk..iblk+31, all 32 heads):
  // one b128 per lane.  ASYNCcnt fence, then workgroup barrier.
  {
    const float* src = W + (size_t)iblk * H_ + threadIdx.x * 4;
    unsigned lds_addr = (unsigned)(uintptr_t)(&wbuf[threadIdx.x * 4]);
    unsigned long long gaddr = (unsigned long long)(uintptr_t)src;
    asm volatile("global_load_async_to_lds_b128 %0, %1, off"
                 :: "v"(lds_addr), "v"(gaddr) : "memory");
    asm volatile("s_wait_asynccnt 0" ::: "memory");
  }
  __syncthreads();

  const int qrow   = ibase + (lane & 15);
  const int khalfA = (lane >> 4) * 16;
  const int jcol   = jbase + (lane & 15);
  const int kselB  = (lane >> 4) * 16;
  const __bf16* bptr = Kb + (size_t)jcol * D_ + kselB;

  // Hoist the 4 B-fragments (K=128) -- invariant across heads.
  v16bf breg[4];
#pragma unroll
  for (int kk = 0; kk < 4; ++kk) breg[kk] = ldB(bptr + kk * 32);

  const int mofs = (lane >> 4) * 8;
  float total[8] = {0.f, 0.f, 0.f, 0.f, 0.f, 0.f, 0.f, 0.f};

  for (int h = 0; h < H_; ++h) {
    const __bf16* aptr = Q + ((size_t)qrow * H_ + h) * D_ + khalfA;
    v8f acc = {};
#pragma unroll
    for (int kk = 0; kk < 4; ++kk) {
      v16bf a = ldA(aptr + kk * 32);
      acc = __builtin_amdgcn_wmma_f32_16x16x32_bf16(
          false, a, false, breg[kk], (short)0, acc, false, false);
    }
#pragma unroll
    for (int v = 0; v < 8; ++v) {
      float sv = acc[v] * inv_sqrt_d;
      sv = sv > 0.f ? sv : 0.f;
      total[v] += sv * wbuf[(wm * 16 + mofs + v) * H_ + h];
    }
  }

#pragma unroll
  for (int v = 0; v < 8; ++v) {
    int i = ibase + mofs + v;
    Sc[(size_t)i * S_LEN + jcol] = (jcol <= i) ? total[v] : -1e30f;
  }
}

// ---------------------------------------------------------------------------
// Top-512 of each 2048-row: LDS bitonic sort (descending) of (key|idx) u64s.
// ---------------------------------------------------------------------------
__global__ __launch_bounds__(256)
void k_topk(const float* __restrict__ Sc,
            float* __restrict__ outS, int* __restrict__ outI) {
  __shared__ unsigned long long keys[S_LEN];
  const int row = blockIdx.x;

  for (int t = threadIdx.x; t < S_LEN; t += 256) {
    unsigned u = __float_as_uint(Sc[(size_t)row * S_LEN + t]);
    unsigned k = (u & 0x80000000u) ? ~u : (u | 0x80000000u);
    keys[t] = ((unsigned long long)k << 32) | (unsigned)t;
  }
  __syncthreads();

  for (int ksz = 2; ksz <= S_LEN; ksz <<= 1) {
    for (int j = ksz >> 1; j > 0; j >>= 1) {
      for (int i = threadIdx.x; i < S_LEN; i += 256) {
        int ixj = i ^ j;
        if (ixj > i) {
          bool descend = ((i & ksz) == 0);
          unsigned long long a = keys[i], b = keys[ixj];
          bool sw = descend ? (a < b) : (a > b);
          if (sw) { keys[i] = b; keys[ixj] = a; }
        }
      }
      __syncthreads();
    }
  }

  for (int t = threadIdx.x; t < TOPK_; t += 256) {
    unsigned long long kv = keys[t];
    unsigned k = (unsigned)(kv >> 32);
    unsigned u = (k & 0x80000000u) ? (k & 0x7FFFFFFFu) : ~k;
    outS[(size_t)row * TOPK_ + t] = __uint_as_float(u);
    outI[(size_t)row * TOPK_ + t] = (int)(kv & 0xFFFFFFFFu);
  }
}

// ---------------------------------------------------------------------------
// Host-side launch
// ---------------------------------------------------------------------------
extern "C" void kernel_launch(void* const* d_in, const int* in_sizes, int n_in,
                              void* d_out, int out_size, void* d_ws, size_t ws_size,
                              hipStream_t stream) {
  const float* hs   = (const float*)d_in[0];  // (1, S, HID)
  const float* cosb = (const float*)d_in[1];  // (1, S, RD/2)
  const float* sinb = (const float*)d_in[2];  // (1, S, RD/2)
  const float* wq   = (const float*)d_in[3];  // (HID, H*D)
  const float* wk   = (const float*)d_in[4];  // (HID, D)
  const float* ww   = (const float*)d_in[5];  // (HID, H)
  (void)in_sizes; (void)n_in; (void)out_size;

  char* ws = (char*)d_ws;
  size_t off = 0;
  auto carve = [&](size_t bytes) -> char* {
    char* p = ws + off;
    off = (off + bytes + 255) & ~(size_t)255;
    return p;
  };
  __bf16* hsb  = (__bf16*)carve((size_t)S_LEN * HID_ * 2);
  __bf16* wqT  = (__bf16*)carve((size_t)NQ_   * HID_ * 2);
  __bf16* wkT  = (__bf16*)carve((size_t)D_    * HID_ * 2);
  __bf16* wwT  = (__bf16*)carve((size_t)H_    * HID_ * 2);
  float*  qf   = (float*) carve((size_t)S_LEN * NQ_  * 4);
  float*  kf   = (float*) carve((size_t)S_LEN * D_   * 4);
  float*  wf   = (float*) carve((size_t)S_LEN * H_   * 4);
  __bf16* qb   = (__bf16*)carve((size_t)S_LEN * NQ_  * 2);
  __bf16* kb   = (__bf16*)carve((size_t)S_LEN * D_   * 2);
  float*  sc   = (float*) carve((size_t)S_LEN * S_LEN * 4);
  (void)ws_size;

  float* outS = (float*)d_out;
  int*   outI = (int*)d_out + (size_t)S_LEN * TOPK_;

  // 1) convert / transpose operands to bf16 (B operands K-contiguous)
  {
    size_t n = (size_t)S_LEN * HID_;
    k_f32_to_bf16<<<dim3((n + 255) / 256), dim3(256), 0, stream>>>(hs, hsb, n);
  }
  {
    size_t n = (size_t)HID_ * NQ_;
    k_transpose_bf16<<<dim3((n + 255) / 256), dim3(256), 0, stream>>>(wq, wqT, HID_, NQ_);
  }
  {
    size_t n = (size_t)HID_ * D_;
    k_transpose_bf16<<<dim3((n + 255) / 256), dim3(256), 0, stream>>>(wk, wkT, HID_, D_);
  }
  {
    size_t n = (size_t)HID_ * H_;
    k_transpose_bf16<<<dim3((n + 255) / 256), dim3(256), 0, stream>>>(ww, wwT, HID_, H_);
  }

  // 2) projections (bf16 WMMA, fp32 accumulate)
  k_wmma_gemm_q<<<dim3(S_LEN / 128, NQ_ / 128), dim3(256), 0, stream>>>(
      hsb, wqT, qf, S_LEN, NQ_, HID_);
  k_wmma_gemm_bf16nt<<<dim3(S_LEN / 32, D_ / 64), dim3(256), 0, stream>>>(
      hsb, wkT, kf, S_LEN, D_, HID_, 1.0f);
  k_wmma_gemm_bf16nt<<<dim3(S_LEN / 32, 1), dim3(256), 0, stream>>>(
      hsb, wwT, wf, S_LEN, H_, HID_, 0.17677669529663687f);  // H^-0.5

  // 3) RoPE + bf16 for the score GEMM
  {
    size_t n = (size_t)S_LEN * H_ * (D_ / 2);
    k_rope_bf16<<<dim3((n + 255) / 256), dim3(256), 0, stream>>>(
        qf, cosb, sinb, qb, H_, D_, RD_);
  }
  {
    size_t n = (size_t)S_LEN * (D_ / 2);
    k_rope_bf16<<<dim3((n + 255) / 256), dim3(256), 0, stream>>>(
        kf, cosb, sinb, kb, 1, D_, RD_);
  }

  // 4) head-reduced causal scores
  k_scores<<<dim3(S_LEN / 32, S_LEN / 64), dim3(256), 0, stream>>>(
      qb, kb, wf, sc, 0.08838834764831845f);  // D^-0.5

  // 5) per-row top-512
  k_topk<<<dim3(S_LEN), dim3(256), 0, stream>>>(sc, outS, outI);
}
